// MultiHeadAttentionWrapper_3616362463617
// MI455X (gfx1250) — matrix operand order repaired and checked by reading, not compiled, not run on hardware
//
#include <hip/hip_runtime.h>
#include <stdint.h>

typedef __attribute__((ext_vector_type(16))) __bf16 v16bf;
typedef __attribute__((ext_vector_type(8)))  __bf16 v8bf;
typedef __attribute__((ext_vector_type(8)))  float  v8f;
typedef __attribute__((ext_vector_type(4)))  float  v4f;

__device__ __forceinline__ v16bf cat16(v8bf lo, v8bf hi) {
    return __builtin_shufflevector(lo, hi, 0,1,2,3,4,5,6,7,8,9,10,11,12,13,14,15);
}

// async global->LDS copy of 16 bytes per lane (ASYNCcnt-tracked, no VGPR data path)
__device__ __forceinline__ void async_copy_b128(uint32_t lds_byte_off, const void* gptr) {
    asm volatile("global_load_async_to_lds_b128 %0, %1, off"
                 :: "v"(lds_byte_off), "v"((uint64_t)(uintptr_t)gptr)
                 : "memory");
}
__device__ __forceinline__ void wait_async0() {
    asm volatile("s_wait_asynccnt 0x0" ::: "memory");
}

// ---------------------------------------------------------------------------
// Kernel 1: QKV projection.  One block = 64 (seq) x 64 (DH) output tile of
// Q, K or V for one (b, h).  4 waves, each wave owns 16 rows x 64 cols.
// WMMA: v_wmma_f32_16x16x32_bf16, K-loop over D=768 in steps of 32.
// Q is pre-scaled by 1/sqrt(DH) = 0.125 on store.
// qkv layout: [mat(3)][B][H][S][DH] bf16.
// ---------------------------------------------------------------------------
__global__ __launch_bounds__(128)
void qkv_proj_kernel(const float* __restrict__ x,
                     const float* __restrict__ Wq,
                     const float* __restrict__ Wk,
                     const float* __restrict__ Wv,
                     __bf16* __restrict__ qkv)
{
    constexpr int BQ = 8, S = 1024, D = 768, H = 12, DH = 64;
    const int m0  = blockIdx.x * 64;
    const int b   = blockIdx.y;
    const int z   = blockIdx.z;
    const int mat = z / H, h = z % H;

    const float* W  = (mat == 0 ? Wq : (mat == 1 ? Wk : Wv)) + (size_t)h * D * DH;
    const float* xb = x + ((size_t)b * S + m0) * D;

    __shared__ __attribute__((aligned(16))) __bf16 Xt[64 * 40]; // [row][k], stride 40
    __shared__ __attribute__((aligned(16))) __bf16 Wt[64 * 40]; // [col][k] (transposed)

    const int  t    = threadIdx.x;
    const int  w    = t >> 5;
    const int  lane = t & 31;
    const int  hl   = lane & 15;
    const bool hiL  = lane >= 16;

    v8f acc[4] = {};

    for (int kk = 0; kk < D; kk += 32) {
        // prefetch next K-step tiles (everything fits in the 192 MB L2)
        if (kk + 32 < D) {
            __builtin_prefetch(&xb[(size_t)(t >> 1) * D + kk + 32 + (t & 1) * 16], 0, 3);
            __builtin_prefetch(&W[(size_t)(kk + 32 + (t >> 2)) * DH + (t & 3) * 16], 0, 3);
        }
        // stage X tile (64x32 f32 -> bf16), float4 loads
#pragma unroll
        for (int i = 0; i < 4; ++i) {
            int chunk = t + i * 128;            // 512 chunks of 4 elements
            int base  = chunk * 4;
            int r = base >> 5, c = base & 31;
            v4f v = *(const v4f*)&xb[(size_t)r * D + kk + c];
#pragma unroll
            for (int e = 0; e < 4; ++e) Xt[r * 40 + c + e] = (__bf16)v[e];
        }
        // stage W tile (32x64 f32) transposed -> Wt[col][k]
#pragma unroll
        for (int i = 0; i < 4; ++i) {
            int chunk = t + i * 128;
            int base  = chunk * 4;
            int r = base >> 6, c = base & 63;
            v4f v = *(const v4f*)&W[(size_t)(kk + r) * DH + c];
#pragma unroll
            for (int e = 0; e < 4; ++e) Wt[(c + e) * 40 + r] = (__bf16)v[e];
        }
        __syncthreads();

        // A fragment (16x32 bf16): lane<16 holds K 0..15, lane>=16 K 16..31
        const int arow = w * 16 + hl;
        const int kb   = hiL ? 16 : 0;
        v8bf alo = *(const v8bf*)&Xt[arow * 40 + kb];
        v8bf ahi = *(const v8bf*)&Xt[arow * 40 + 8 + kb];
        v16bf a  = cat16(alo, ahi);

        // batch all 4 B fragments, then issue the 4 WMMAs back-to-back
        const int kb2 = hiL ? 16 : 0;  // B fragment: lane<16 -> K 0..15, lane>=16 -> K 16..31
        v16bf bbf[4];
#pragma unroll
        for (int nt = 0; nt < 4; ++nt) {
            int col = nt * 16 + hl;
            bbf[nt] = cat16(*(const v8bf*)&Wt[col * 40 + kb2],
                            *(const v8bf*)&Wt[col * 40 + kb2 + 8]);
        }
#pragma unroll
        for (int nt = 0; nt < 4; ++nt)
            acc[nt] = __builtin_amdgcn_wmma_f32_16x16x32_bf16(
                false, a, false, bbf[nt], (short)0, acc[nt], false, false);
        __syncthreads();
    }

    const float sc = (mat == 0) ? 0.125f : 1.0f;   // fold 1/sqrt(DH) into Q
    __bf16* outp = qkv + ((((size_t)mat * BQ + b) * H + h) * S + m0) * DH;
#pragma unroll
    for (int nt = 0; nt < 4; ++nt)
#pragma unroll
        for (int r = 0; r < 8; ++r) {
            int row = w * 16 + (hiL ? 8 : 0) + r;  // C layout: lanes 16..31 hold M=8..15
            int col = nt * 16 + hl;
            outp[(size_t)row * DH + col] = (__bf16)(acc[nt][r] * sc);
        }
}

// ---------------------------------------------------------------------------
// Kernel 2: causal flash attention.  One block = 64 query rows for one (b,h).
// 4 waves, wave w owns rows m0+16w .. m0+16w+15, full DH=64 accumulator.
// K tile staged with GLOBAL_LOAD_ASYNC_TO_LDS_B128 (ASYNCcnt), V staged
// transposed through VGPRs.  Per 32-key tile: 4 WMMA (Q.K^T) + online
// softmax + 4 WMMA (P.V).
// ---------------------------------------------------------------------------
__global__ __launch_bounds__(128)
void flash_attn_kernel(const __bf16* __restrict__ qkv, float* __restrict__ out)
{
    constexpr int BQ = 8, S = 1024, H = 12, DH = 64;
    const size_t headsz = (size_t)S * DH;
    const size_t matsz  = (size_t)BQ * H * headsz;
    const int m0 = blockIdx.x * 64;
    const int b  = blockIdx.y, h = blockIdx.z;
    const size_t headoff = ((size_t)b * H + h) * headsz;

    const __bf16* Q  = qkv + headoff + (size_t)m0 * DH;       // already scaled
    const __bf16* Kp = qkv + matsz + headoff;
    const __bf16* Vp = qkv + 2 * matsz + headoff;

    __shared__ __attribute__((aligned(16))) __bf16 Kt[32 * 72];     // [key][dh], stride 72
    __shared__ __attribute__((aligned(16))) __bf16 Vt[64 * 40];     // [dh][key] (transposed)
    __shared__ __attribute__((aligned(16))) __bf16 Pt[4][16 * 40];  // per-wave P tile

    const int  t    = threadIdx.x;
    const int  w    = t >> 5;
    const int  lane = t & 31;
    const int  hl   = lane & 15;
    const bool hiL  = lane >= 16;
    const int  kb   = hiL ? 16 : 0;

    // Q fragments for the two 32-wide DH chunks
    v16bf qa[2];
#pragma unroll
    for (int c = 0; c < 2; ++c) {
        const __bf16* qr = &Q[(size_t)(w * 16 + hl) * DH + c * 32];
        qa[c] = cat16(*(const v8bf*)&qr[kb], *(const v8bf*)&qr[8 + kb]);
    }

    v8f   o[4] = {};
    float m8[8], l8[8];
#pragma unroll
    for (int r = 0; r < 8; ++r) { m8[r] = -3.0e38f; l8[r] = 0.0f; }

    const int rowbase = m0 + w * 16 + (hiL ? 8 : 0);
    const int kend    = m0 + 64;

    // per-thread chunk coordinates for staging (256 chunks of 8 bf16)
    const int r0 = (t * 8) >> 6,  c0 = (t * 8) & 63;          // chunk t
    const int r1 = ((t + 128) * 8) >> 6, c1 = ((t + 128) * 8) & 63; // chunk t+128

    for (int j0 = 0; j0 < kend; j0 += 32) {
        // ---- async K tile stage: global -> LDS, no VGPR round trip ----
        async_copy_b128((uint32_t)(uintptr_t)&Kt[r0 * 72 + c0],
                        &Kp[(size_t)(j0 + r0) * DH + c0]);
        async_copy_b128((uint32_t)(uintptr_t)&Kt[r1 * 72 + c1],
                        &Kp[(size_t)(j0 + r1) * DH + c1]);
        // ---- V tile staged transposed (needs the shuffle, so via VGPRs) ----
#pragma unroll
        for (int i = 0; i < 2; ++i) {
            int r = i ? r1 : r0, c = i ? c1 : c0;
            v8bf vv = *(const v8bf*)&Vp[(size_t)(j0 + r) * DH + c];
#pragma unroll
            for (int e = 0; e < 8; ++e) Vt[(c + e) * 40 + r] = vv[e];
        }
        // prefetch next tile while this one is consumed
        if (j0 + 32 < kend) {
            __builtin_prefetch(&Kp[(size_t)(j0 + 32 + r0) * DH + c0], 0, 3);
            __builtin_prefetch(&Vp[(size_t)(j0 + 32 + r0) * DH + c0], 0, 3);
        }
        wait_async0();
        __syncthreads();

        if (j0 <= m0 + w * 16) {  // wave-uniform: tile has at least one live key per row
            // ---- scores S = Q.K^T: batch B-fragment loads, then 4 wmma ----
            v16bf kf[2][2];
#pragma unroll
            for (int c = 0; c < 2; ++c)
#pragma unroll
                for (int ns = 0; ns < 2; ++ns) {
                    const __bf16* kr = &Kt[(ns * 16 + hl) * 72 + c * 32 + (hiL ? 16 : 0)];
                    kf[c][ns] = cat16(*(const v8bf*)&kr[0], *(const v8bf*)&kr[8]);
                }
            v8f s[2] = {};
#pragma unroll
            for (int c = 0; c < 2; ++c)
#pragma unroll
                for (int ns = 0; ns < 2; ++ns)
                    s[ns] = __builtin_amdgcn_wmma_f32_16x16x32_bf16(
                        false, qa[c], false, kf[c][ns], (short)0, s[ns], false, false);

            // ---- causal mask + online softmax ----
#pragma unroll
            for (int r = 0; r < 8; ++r) {
                int row = rowbase + r;
#pragma unroll
                for (int ns = 0; ns < 2; ++ns) {
                    int key = j0 + ns * 16 + hl;
                    if (key > row) s[ns][r] = -1.0e30f;
                }
                float mx = fmaxf(s[0][r], s[1][r]);
#pragma unroll
                for (int off = 1; off < 16; off <<= 1)
                    mx = fmaxf(mx, __shfl_xor(mx, off, 32));   // stays within 16-lane half
                float mnew = fmaxf(m8[r], mx);
                float corr = __expf(m8[r] - mnew);
                float p0 = __expf(s[0][r] - mnew);
                float p1 = __expf(s[1][r] - mnew);
                s[0][r] = p0; s[1][r] = p1;
                float rs = p0 + p1;
#pragma unroll
                for (int off = 1; off < 16; off <<= 1)
                    rs += __shfl_xor(rs, off, 32);
                l8[r] = l8[r] * corr + rs;
                m8[r] = mnew;
#pragma unroll
                for (int nt = 0; nt < 4; ++nt) o[nt][r] *= corr;
            }

            // ---- transpose P (C layout -> A layout) through per-wave LDS ----
#pragma unroll
            for (int r = 0; r < 8; ++r) {
                int rl = (hiL ? 8 : 0) + r;
                Pt[w][rl * 40 + hl]      = (__bf16)s[0][r];
                Pt[w][rl * 40 + 16 + hl] = (__bf16)s[1][r];
            }
            asm volatile("s_wait_dscnt 0" ::: "memory");  // wave-local LDS RAW fence
            v16bf pa = cat16(*(const v8bf*)&Pt[w][hl * 40 + kb],
                             *(const v8bf*)&Pt[w][hl * 40 + 8 + kb]);

            // ---- O += P.V: batch B-fragment loads, then 4 wmma ----
            v16bf vf[4];
#pragma unroll
            for (int nt = 0; nt < 4; ++nt) {
                const __bf16* vr = &Vt[(nt * 16 + hl) * 40 + (hiL ? 16 : 0)];
                vf[nt] = cat16(*(const v8bf*)&vr[0], *(const v8bf*)&vr[8]);
            }
#pragma unroll
            for (int nt = 0; nt < 4; ++nt)
                o[nt] = __builtin_amdgcn_wmma_f32_16x16x32_bf16(
                    false, pa, false, vf[nt], (short)0, o[nt], false, false);
        }
        __syncthreads();
    }

    // normalize and write ctx -> out[b][s][h*64+dh] (fp32)
#pragma unroll
    for (int nt = 0; nt < 4; ++nt)
#pragma unroll
        for (int r = 0; r < 8; ++r) {
            int srow = m0 + w * 16 + (hiL ? 8 : 0) + r;
            int col  = h * DH + nt * 16 + hl;
            out[((size_t)b * S + srow) * (H * DH) + col] = o[nt][r] / l8[r];
        }
}

// ---------------------------------------------------------------------------
extern "C" void kernel_launch(void* const* d_in, const int* in_sizes, int n_in,
                              void* d_out, int out_size, void* d_ws, size_t ws_size,
                              hipStream_t stream)
{
    (void)in_sizes; (void)n_in; (void)out_size; (void)ws_size;
    const float* x  = (const float*)d_in[0];
    const float* Wq = (const float*)d_in[1];
    const float* Wk = (const float*)d_in[2];
    const float* Wv = (const float*)d_in[3];
    float* out = (float*)d_out;

    // workspace: Q,K,V in bf16, layout [3][B][H][S][DH] = 36 MB
    __bf16* qkv = (__bf16*)d_ws;

    dim3 blk(128, 1, 1);
    qkv_proj_kernel<<<dim3(16, 8, 36), blk, 0, stream>>>(x, Wq, Wk, Wv, qkv);
    flash_attn_kernel<<<dim3(16, 8, 12), blk, 0, stream>>>(qkv, out);
}
